// MessagePassingLayer_85358180041081
// MI455X (gfx1250) — matrix operand order repaired and checked by reading, not compiled, not run on hardware
//
#include <hip/hip_runtime.h>

#define C_IN 128
#define C_OUT 128
#define ATT_H 16
#define HH 256
#define WW 256
#define HW 65536
#define BB 2
#define NPIX (BB*HW)
#define PT 128            // pixels per workgroup tile
#define NWG (NPIX/PT)     // 1024

// fused first GEMM rows: [Wn(128); We(128); Wa1_self(16); Wa1_shift(16)] = 288
#define M1 288
#define MT1 (M1/16)       // 18
#define KT1 (C_IN/32)     // 4
#define MT2 (C_OUT/16)    // 8
#define KT2 (2*C_OUT/32)  // 8

#define XS_STRIDE 136     // bf16 per row in k1 LDS tile (272B: 16B-aligned, conflict-free)
#define CB_STRIDE 264     // bf16 per row in k2 LDS tile (528B: 16B-aligned, conflict-free)

typedef __attribute__((ext_vector_type(16))) __bf16 v16bf;
typedef __attribute__((ext_vector_type(8)))  float  v8f;

union Frag {
    uint4    q[2];
    unsigned u[8];
    v16bf    v;
};

__device__ __forceinline__ unsigned short f2bf(float f) {
    unsigned u = __float_as_uint(f);
    u += 0x7FFFu + ((u >> 16) & 1u);    // round-to-nearest-even
    return (unsigned short)(u >> 16);
}
__device__ __forceinline__ float bf2f(unsigned short h) {
    return __uint_as_float(((unsigned)h) << 16);
}

// gfx1250 async global->LDS copy (VGLOBAL GLOBAL_LOAD_ASYNC_TO_LDS_B128, op 98):
// VDST = LDS byte offset VGPR, VADDR = 64-bit global address, tracked by ASYNCcnt.
__device__ __forceinline__ void async_ld_b128(unsigned lds_off, const void* gptr) {
    asm volatile("global_load_async_to_lds_b128 %0, %1, off"
                 :: "v"(lds_off), "v"(gptr) : "memory");
}
__device__ __forceinline__ void wait_async0() {
    asm volatile("s_wait_asynccnt 0x0" ::: "memory");
}

// ---------------------------------------------------------------------------
// Kernel 0: pack weights into WMMA A-fragment layout (16-bit A 16x32):
// lane l: m=l&15, hk=l>>4; VGPR j holds K pair k0,k0+1 with
// k0 = 2j + 16*hk    (contiguous k order, matches B row reads)
// ---------------------------------------------------------------------------
__global__ void k_pack(const float* __restrict__ Wn, const float* __restrict__ We,
                       const float* __restrict__ Wa1,
                       const float* __restrict__ bn, const float* __restrict__ be,
                       const float* __restrict__ ba1,
                       const float* __restrict__ Wu,
                       unsigned* __restrict__ wallp, unsigned* __restrict__ wup,
                       float* __restrict__ ball)
{
    int tid = blockIdx.x * 256 + threadIdx.x;
    if (tid < MT1*KT1*32) {
        int lane = tid & 31, f = tid >> 5;
        int mt = f / KT1, kt = f % KT1;
        int m = lane & 15, hk = lane >> 4;
        int row = mt*16 + m;
        const float* src;
        if (row < 128)       src = Wn + (size_t)row*C_IN;
        else if (row < 256)  src = We + (size_t)(row-128)*C_IN;
        else if (row < 272)  src = Wa1 + (size_t)(row-256)*(2*C_IN);        // self half
        else                 src = Wa1 + (size_t)(row-272)*(2*C_IN) + C_IN; // shift half
        #pragma unroll
        for (int j = 0; j < 8; ++j) {
            int k0 = 2*j + 16*hk;
            int kg = kt*32 + k0;
            unsigned lo = f2bf(src[kg]);
            unsigned hi = f2bf(src[kg+1]);
            wallp[(size_t)f*256 + lane*8 + j] = lo | (hi << 16);
        }
    } else if (tid < MT1*KT1*32 + MT2*KT2*32) {
        int t = tid - MT1*KT1*32;
        int lane = t & 31, f = t >> 5;
        int kt = f % KT2;
        int m = lane & 15, hk = lane >> 4;
        int row = (f / KT2)*16 + m;
        const float* src = Wu + (size_t)row*(2*C_OUT);
        #pragma unroll
        for (int j = 0; j < 8; ++j) {
            int k0 = 2*j + 16*hk;
            int kg = kt*32 + k0;
            unsigned lo = f2bf(src[kg]);
            unsigned hi = f2bf(src[kg+1]);
            wup[(size_t)f*256 + lane*8 + j] = lo | (hi << 16);
        }
    } else {
        int t = tid - (MT1*KT1*32 + MT2*KT2*32);
        if (t < M1) {
            float v;
            if (t < 128)      v = bn[t];
            else if (t < 256) v = be[t-128];
            else if (t < 272) v = ba1[t-256];  // ba1 folded into p
            else              v = 0.f;         // q carries no bias
            ball[t] = v;
        }
    }
}

// ---------------------------------------------------------------------------
// Kernel 1: D[288, pix] = Wall @ x.
// node/edge written PIXEL-MAJOR bf16 ([b][s][c], 256B rows) via LDS transpose
// so kernel 2 can async-copy rows straight into LDS. p/q stay channel-major f32.
// ---------------------------------------------------------------------------
__global__ void __launch_bounds__(256) k_gemm1(
    const float* __restrict__ x, const unsigned* __restrict__ wallp,
    const float* __restrict__ ball,
    unsigned short* __restrict__ node, unsigned short* __restrict__ edge,
    float* __restrict__ pbuf, float* __restrict__ qbuf)
{
    __shared__ unsigned short xs[PT*XS_STRIDE];   // 34 816 B, reused as transpose buf
    int tid = threadIdx.x;
    int pix0 = blockIdx.x * PT;
    int b = pix0 >> 16;          // HW = 65536, tiles never straddle batches
    int s0 = pix0 & (HW-1);
    const float* xb = x + (size_t)b*C_IN*HW + s0;

    // stage tile, f32 -> bf16, [pixel][channel]
    for (int e = tid; e < C_IN*PT; e += 256) {
        int c = e >> 7;
        int px = e & (PT-1);
        xs[px*XS_STRIDE + c] = f2bf(xb[(size_t)c*HW + px]);
    }
    __syncthreads();

    int lane = tid & 31;
    int wave = tid >> 5;         // 8 waves = 8 N-tiles of 16 pixels
    int n = lane & 15, hk = lane >> 4;
    int px = wave*16 + n;

    // B fragments into registers (xs becomes free afterwards)
    Frag bfr[KT1];
    {
        const char* rowp = (const char*)xs + (size_t)px*(XS_STRIDE*2);
        #pragma unroll
        for (int kt = 0; kt < KT1; ++kt) {
            const uint4* p4 = (const uint4*)(rowp + kt*64 + hk*32);
            bfr[kt].q[0] = p4[0];
            bfr[kt].q[1] = p4[1];
        }
    }
    __syncthreads();   // everyone done reading xs before it is overwritten

    int s = s0 + px;

    // ---- phase 1: node (mt 0..7), transpose via LDS, coalesced store ----
    for (int mt = 0; mt < 8; ++mt) {
        v8f acc;
        #pragma unroll
        for (int r = 0; r < 8; ++r) acc[r] = ball[mt*16 + r + 8*hk];
        Frag a[KT1];
        #pragma unroll
        for (int kt = 0; kt < KT1; ++kt) {      // one clause of 8x b128
            const uint4* p4 = (const uint4*)(wallp + ((size_t)(mt*KT1 + kt)*32 + lane)*8);
            a[kt].q[0] = p4[0]; a[kt].q[1] = p4[1];
        }
        #pragma unroll
        for (int kt = 0; kt < KT1; ++kt)
            acc = __builtin_amdgcn_wmma_f32_16x16x32_bf16(false, a[kt].v, false, bfr[kt].v,
                                                          (short)0, acc, false, false);
        #pragma unroll
        for (int r = 0; r < 8; ++r)
            xs[px*XS_STRIDE + mt*16 + r + 8*hk] = f2bf(acc[r]);
    }
    __syncthreads();
    {
        unsigned short* dst = node + (size_t)b*HW*C_OUT + (size_t)s0*C_OUT;
        for (int e = tid; e < PT*16; e += 256) {      // 2048 x 16B chunks
            int p = e >> 4, c16 = e & 15;
            uint4 v = *(const uint4*)((const char*)xs + (size_t)p*(XS_STRIDE*2) + c16*16);
            *(uint4*)((char*)dst + (size_t)p*256 + c16*16) = v;
        }
    }
    __syncthreads();

    // ---- phase 2: edge (mt 8..15), same transpose path ----
    for (int mt = 8; mt < 16; ++mt) {
        v8f acc;
        #pragma unroll
        for (int r = 0; r < 8; ++r) acc[r] = ball[mt*16 + r + 8*hk];
        Frag a[KT1];
        #pragma unroll
        for (int kt = 0; kt < KT1; ++kt) {
            const uint4* p4 = (const uint4*)(wallp + ((size_t)(mt*KT1 + kt)*32 + lane)*8);
            a[kt].q[0] = p4[0]; a[kt].q[1] = p4[1];
        }
        #pragma unroll
        for (int kt = 0; kt < KT1; ++kt)
            acc = __builtin_amdgcn_wmma_f32_16x16x32_bf16(false, a[kt].v, false, bfr[kt].v,
                                                          (short)0, acc, false, false);
        #pragma unroll
        for (int r = 0; r < 8; ++r)
            xs[px*XS_STRIDE + (mt-8)*16 + r + 8*hk] = f2bf(acc[r]);
    }
    __syncthreads();
    {
        unsigned short* dst = edge + (size_t)b*HW*C_OUT + (size_t)s0*C_OUT;
        for (int e = tid; e < PT*16; e += 256) {
            int p = e >> 4, c16 = e & 15;
            uint4 v = *(const uint4*)((const char*)xs + (size_t)p*(XS_STRIDE*2) + c16*16);
            *(uint4*)((char*)dst + (size_t)p*256 + c16*16) = v;
        }
    }

    // ---- phase 3: p / q (mt 16, 17), channel-major f32 ----
    for (int mt = 16; mt < MT1; ++mt) {
        v8f acc;
        #pragma unroll
        for (int r = 0; r < 8; ++r) acc[r] = ball[mt*16 + r + 8*hk];
        Frag a[KT1];
        #pragma unroll
        for (int kt = 0; kt < KT1; ++kt) {
            const uint4* p4 = (const uint4*)(wallp + ((size_t)(mt*KT1 + kt)*32 + lane)*8);
            a[kt].q[0] = p4[0]; a[kt].q[1] = p4[1];
        }
        #pragma unroll
        for (int kt = 0; kt < KT1; ++kt)
            acc = __builtin_amdgcn_wmma_f32_16x16x32_bf16(false, a[kt].v, false, bfr[kt].v,
                                                          (short)0, acc, false, false);
        float* dst = (mt == 16 ? pbuf : qbuf) + (size_t)b*ATT_H*HW + s;
        #pragma unroll
        for (int r = 0; r < 8; ++r)
            dst[(size_t)(r + 8*hk)*HW] = acc[r];
    }
}

// ---------------------------------------------------------------------------
// Kernel 2: async node->LDS copy overlapped with attention, aggregate shifted
// edges, Wu GEMM (WMMA), partial GroupNorm stats.
// ---------------------------------------------------------------------------
__global__ void __launch_bounds__(256) k_gemm2(
    const unsigned short* __restrict__ node, const unsigned short* __restrict__ edge,
    const float* __restrict__ pbuf, const float* __restrict__ qbuf,
    const unsigned* __restrict__ wup, const float* __restrict__ bu,
    const float* __restrict__ Wa2, const float* __restrict__ ba2,
    float* __restrict__ outNN, float* __restrict__ partials)
{
    extern __shared__ char sm[];
    unsigned short* comb = (unsigned short*)sm;           // PT * CB_STRIDE bf16 (offset 0)
    float* att = (float*)(sm + PT*CB_STRIDE*2);           // 4 * PT
    float* red = att + 4*PT;                              // 512 floats

    int tid = threadIdx.x;
    int pix0 = blockIdx.x * PT;
    int b = pix0 >> 16;
    int s0 = pix0 & (HW-1);

    // ---- issue async copies: node rows (pixel-major, 256B) -> comb[:,0:128] ----
    {
        const unsigned short* nsrc = node + (size_t)b*HW*C_OUT + (size_t)s0*C_OUT;
        for (int e = tid; e < PT*16; e += 256) {          // 8 async issues per lane
            int px = e >> 4, c16 = e & 15;
            unsigned lds_off = (unsigned)(px*(CB_STRIDE*2) + c16*16);
            async_ld_b128(lds_off, (const char*)nsrc + (size_t)px*256 + c16*16);
        }
    }

    // ---- attention for 4 directions (overlaps with async copy) ----
    if (tid < PT) {
        int s = s0 + tid;
        int h = s >> 8, w = s & (WW-1);
        const float* pb = pbuf + (size_t)b*ATT_H*HW + s;
        float pv[ATT_H];
        #pragma unroll
        for (int j = 0; j < ATT_H; ++j) pv[j] = pb[(size_t)j*HW];
        const int dS[4]   = { -WW, WW, -1, 1 };
        const bool vld[4] = { h > 0, h < HH-1, w > 0, w < WW-1 };
        float b2 = ba2[0];
        #pragma unroll
        for (int d = 0; d < 4; ++d) {
            float a = 0.f;
            if (vld[d]) {
                const float* qb = qbuf + (size_t)b*ATT_H*HW + (s + dS[d]);
                float dot = b2;
                #pragma unroll
                for (int j = 0; j < ATT_H; ++j) {
                    float hv = pv[j] + qb[(size_t)j*HW];
                    hv = hv > 0.f ? hv : 0.2f*hv;     // LeakyReLU(0.2)
                    dot += Wa2[j]*hv;
                }
                a = 1.f/(1.f + __expf(-dot));         // sigmoid
            }
            att[d*PT + tid] = a;
        }
    }
    wait_async0();
    __syncthreads();

    // ---- aggr -> comb[:, 128:256]; lanes walk channels (coalesced 256B rows) ----
    {
        const unsigned short* edgeb = edge + (size_t)b*HW*C_OUT;
        for (int e = tid; e < C_OUT*PT; e += 256) {
            int c  = e & 127;
            int px = e >> 7;
            int s = s0 + px;
            int h = s >> 8, w = s & (WW-1);
            const int dS[4]   = { -WW, WW, -1, 1 };
            const bool vld[4] = { h > 0, h < HH-1, w > 0, w < WW-1 };
            float a = 0.f;
            #pragma unroll
            for (int d = 0; d < 4; ++d)
                if (vld[d])
                    a += bf2f(edgeb[(size_t)(s + dS[d])*C_OUT + c]) * att[d*PT + px];
            comb[px*CB_STRIDE + C_OUT + c] = f2bf(a);
        }
    }
    __syncthreads();

    // ---- new_nodes = Wu @ [node; aggr] + bu ----
    int lane = tid & 31, wave = tid >> 5;
    int n = lane & 15, hk = lane >> 4;
    int px = wave*16 + n;

    Frag bfr[KT2];
    const char* rowp = (const char*)comb + (size_t)px*(CB_STRIDE*2);
    #pragma unroll
    for (int kt = 0; kt < KT2; ++kt) {
        const uint4* p4 = (const uint4*)(rowp + kt*64 + hk*32);
        bfr[kt].q[0] = p4[0]; bfr[kt].q[1] = p4[1];
    }

    float ls = 0.f, ls2 = 0.f;
    int s = s0 + px;
    for (int mt = 0; mt < MT2; ++mt) {
        v8f acc;
        #pragma unroll
        for (int r = 0; r < 8; ++r) acc[r] = bu[mt*16 + r + 8*hk];
        #pragma unroll
        for (int g = 0; g < 2; ++g) {             // two groups of 4 frags
            Frag a[4];
            #pragma unroll
            for (int k = 0; k < 4; ++k) {
                int kt = g*4 + k;
                const uint4* p4 = (const uint4*)(wup + ((size_t)(mt*KT2 + kt)*32 + lane)*8);
                a[k].q[0] = p4[0]; a[k].q[1] = p4[1];
            }
            #pragma unroll
            for (int k = 0; k < 4; ++k)
                acc = __builtin_amdgcn_wmma_f32_16x16x32_bf16(false, a[k].v, false,
                                                              bfr[g*4+k].v,
                                                              (short)0, acc, false, false);
        }
        #pragma unroll
        for (int r = 0; r < 8; ++r) {
            int ch = mt*16 + r + 8*hk;
            float v = acc[r];
            outNN[(size_t)b*C_OUT*HW + (size_t)ch*HW + s] = v;
            ls += v; ls2 += v*v;
        }
    }

    // deterministic per-block reduction of sum / sumsq
    red[tid] = ls; red[256 + tid] = ls2;
    __syncthreads();
    for (int off = 128; off > 0; off >>= 1) {
        if (tid < off) { red[tid] += red[tid+off]; red[256+tid] += red[256+tid+off]; }
        __syncthreads();
    }
    if (tid == 0) {
        partials[blockIdx.x*2]     = red[0];
        partials[blockIdx.x*2 + 1] = red[256];
    }
}

// ---------------------------------------------------------------------------
// Kernel 3: reduce partials -> per-batch mean and rsqrt(var+eps)
// ---------------------------------------------------------------------------
__global__ void k_stats(const float* __restrict__ partials, float* __restrict__ stat)
{
    __shared__ float red[512];
    int tid = threadIdx.x;
    for (int b = 0; b < BB; ++b) {
        float s = 0.f, s2 = 0.f;
        for (int i = tid; i < NWG/BB; i += 256) {
            s  += partials[(b*(NWG/BB) + i)*2];
            s2 += partials[(b*(NWG/BB) + i)*2 + 1];
        }
        red[tid] = s; red[256+tid] = s2;
        __syncthreads();
        for (int off = 128; off > 0; off >>= 1) {
            if (tid < off) { red[tid] += red[tid+off]; red[256+tid] += red[256+tid+off]; }
            __syncthreads();
        }
        if (tid == 0) {
            float N = (float)((size_t)C_OUT*HW);
            float mu = red[0]/N;
            float var = red[256]/N - mu*mu;
            stat[b*2]   = mu;
            stat[b*2+1] = rsqrtf(var + 1e-5f);
        }
        __syncthreads();
    }
}

// ---------------------------------------------------------------------------
// Kernel 4: GroupNorm affine + ReLU + residual, in place on d_out
// ---------------------------------------------------------------------------
__global__ void __launch_bounds__(256) k_norm(
    float* __restrict__ outNN, const float* __restrict__ x,
    const float* __restrict__ gamma, const float* __restrict__ beta,
    const float* __restrict__ stat)
{
    size_t total = (size_t)BB*C_OUT*HW;
    for (size_t i = blockIdx.x*256ull + threadIdx.x; i < total; i += (size_t)gridDim.x*256ull) {
        int b = (int)(i >> 23);            // 128*65536 = 2^23 per batch
        int c = (int)(i >> 16) & 127;
        float v = (outNN[i] - stat[b*2]) * stat[b*2+1] * gamma[c] + beta[c];
        v = v > 0.f ? v : 0.f;
        outNN[i] = v + x[i];
    }
}

// ---------------------------------------------------------------------------
extern "C" void kernel_launch(void* const* d_in, const int* in_sizes, int n_in,
                              void* d_out, int out_size, void* d_ws, size_t ws_size,
                              hipStream_t stream)
{
    const float* x     = (const float*)d_in[0];
    const float* Wn    = (const float*)d_in[1];
    const float* bn    = (const float*)d_in[2];
    const float* We    = (const float*)d_in[3];
    const float* be    = (const float*)d_in[4];
    const float* Wa1   = (const float*)d_in[5];
    const float* ba1   = (const float*)d_in[6];
    const float* Wa2   = (const float*)d_in[7];
    const float* ba2   = (const float*)d_in[8];
    const float* Wu    = (const float*)d_in[9];
    const float* bu    = (const float*)d_in[10];
    const float* gamma = (const float*)d_in[11];
    const float* beta  = (const float*)d_in[12];

    char* ws = (char*)d_ws;
    const size_t OFF_WALL = 0;                                       // 73 728 B
    const size_t OFF_WU   = OFF_WALL + (size_t)MT1*KT1*32*8*4;
    const size_t OFF_BALL = OFF_WU   + (size_t)MT2*KT2*32*8*4;       // 65 536 B
    const size_t OFF_PART = (OFF_BALL + (size_t)M1*4 + 255) & ~(size_t)255;
    const size_t OFF_STAT = OFF_PART + (size_t)NWG*2*4;
    const size_t OFF_NODE = (OFF_STAT + 16 + 255) & ~(size_t)255;
    const size_t OFF_EDGE = OFF_NODE + (size_t)BB*C_OUT*HW*2;        // bf16 pixel-major
    const size_t OFF_P    = OFF_EDGE + (size_t)BB*C_OUT*HW*2;        // bf16 pixel-major
    const size_t OFF_Q    = OFF_P    + (size_t)BB*ATT_H*HW*4;        // f32

    unsigned* wallp        = (unsigned*)(ws + OFF_WALL);
    unsigned* wup          = (unsigned*)(ws + OFF_WU);
    float*    ball         = (float*)(ws + OFF_BALL);
    float*    partials     = (float*)(ws + OFF_PART);
    float*    stat         = (float*)(ws + OFF_STAT);
    unsigned short* nodeb  = (unsigned short*)(ws + OFF_NODE);
    unsigned short* edgeb  = (unsigned short*)(ws + OFF_EDGE);
    float*    pbuf         = (float*)(ws + OFF_P);
    float*    qbuf         = (float*)(ws + OFF_Q);
    float*    outNN        = (float*)d_out;

    int packThreads = MT1*KT1*32 + MT2*KT2*32 + M1;
    k_pack<<<(packThreads + 255)/256, 256, 0, stream>>>(Wn, We, Wa1, bn, be, ba1, Wu,
                                                        wallp, wup, ball);
    k_gemm1<<<NWG, 256, 0, stream>>>(x, wallp, ball, nodeb, edgeb, pbuf, qbuf);
    const size_t smem2 = (size_t)PT*CB_STRIDE*2 + 4*PT*4 + 512*4;    // 71 680 B
    k_gemm2<<<NWG, 256, smem2, stream>>>(nodeb, edgeb, pbuf, qbuf, wup, bu,
                                         Wa2, ba2, outNN, partials);
    k_stats<<<1, 256, 0, stream>>>(partials, stat);
    k_norm<<<2048, 256, 0, stream>>>(outNN, x, gamma, beta, stat);
}
